// plQA_59760174956704
// MI455X (gfx1250) — compile-verified, matrix operand order checked by reading, not run on hardware
//
#include <hip/hip_runtime.h>

constexpr int BB = 8;
constexpr int TT = 4096;
constexpr int CC = 1024;
constexpr int QQ = 512;

typedef __attribute__((ext_vector_type(16))) _Float16 v16h;
typedef __attribute__((ext_vector_type(8)))  _Float16 v8h;
typedef __attribute__((ext_vector_type(8)))  float    v8f;
typedef __attribute__((ext_vector_type(4)))  int      v4i;

#define LN10000 9.2103403719761836f

// ---------------------------------------------------------------------------
// Async global->LDS (CDNA5 GLOBAL_LOAD_ASYNC_TO_LDS_B128, ASYNCcnt-tracked),
// with a synchronous fallback if the builtin is absent on this toolchain.
// Probe result: the builtin takes int4* pointers (generic AS), 4 args.
// ---------------------------------------------------------------------------
#if __has_builtin(__builtin_amdgcn_global_load_async_to_lds_b128)
#define HAS_ASYNC_LDS 1
#else
#define HAS_ASYNC_LDS 0
#endif

static __device__ inline void cp16_async(_Float16* dst, const _Float16* src) {
#if HAS_ASYNC_LDS
  __builtin_amdgcn_global_load_async_to_lds_b128((v4i*)src, (v4i*)dst, 0, 0);
#else
  *(v8h*)dst = *(const v8h*)src;
#endif
}

static __device__ inline void async_wait0() {
#if HAS_ASYNC_LDS
#if __has_builtin(__builtin_amdgcn_s_wait_asynccnt)
  __builtin_amdgcn_s_wait_asynccnt(0);
#else
  asm volatile("s_wait_asynccnt 0x0" ::: "memory");
#endif
#endif
}

// Load a 16-half WMMA fragment for one lane from an LDS row.
// Fragment layout (16x16x32 f16): lane<16 -> k in {0..7, 16..23}; lane>=16 -> k in {8..15, 24..31}.
static __device__ inline v16h frag_ld(const _Float16* rowp, int k0) {
  v8h lo = *(const v8h*)(rowp + k0);
  v8h hi = *(const v8h*)(rowp + k0 + 16);
  v16h r;
#pragma unroll
  for (int i = 0; i < 8; ++i) { r[i] = lo[i]; r[i + 8] = hi[i]; }
  return r;
}

static __device__ inline v8f wmma_f16(v16h a, v16h b, v8f c) {
  return __builtin_amdgcn_wmma_f32_16x16x32_f16(false, a, false, b, (short)0, c, false, false);
}

// ---------------------------------------------------------------------------
// Kernel 0: f32 -> f16 bulk convert (x, Wk, Wv, Wf) so every later stage moves
// half the bytes and can use raw async B128 copies into LDS.
// ---------------------------------------------------------------------------
__global__ __launch_bounds__(256) void cvt_f16_kernel(const float* __restrict__ src,
                                                      _Float16* __restrict__ dst, int n8) {
  int i = blockIdx.x * blockDim.x + threadIdx.x;
  if (i >= n8) return;
  const float* s = src + (size_t)i * 8;
  v8h h;
#pragma unroll
  for (int e = 0; e < 8; ++e) h[e] = (_Float16)s[e];
  *(v8h*)(dst + (size_t)i * 8) = h;
}

// ---------------------------------------------------------------------------
// Kernel 1: Q = rope(W_Q) -> f16   (QS x C)
// ---------------------------------------------------------------------------
__global__ __launch_bounds__(256) void rope_q_kernel(const float* __restrict__ Wq,
                                                     _Float16* __restrict__ Qh) {
  int p = blockIdx.x * blockDim.x + threadIdx.x;   // pair index
  if (p >= QQ * (CC / 2)) return;
  int q  = p / (CC / 2);
  int i  = p - q * (CC / 2);
  int c0 = 2 * i;
  float x0 = Wq[q * CC + c0];
  float x1 = Wq[q * CC + c0 + 1];
  float invf  = __expf(-LN10000 * (float)c0 * (1.0f / CC));
  float theta = (float)q * invf;
  float s, cz;
  __sincosf(theta, &s, &cz);
  Qh[q * CC + c0]     = (_Float16)(x0 * cz - x1 * s);
  Qh[q * CC + c0 + 1] = (_Float16)(x1 * cz + x0 * s);
}

// ---------------------------------------------------------------------------
// Kernel 2: K = rope(x@Wk.T + bk) -> Kh[b*T+t][c] (f16)
//           V = x@Wv.T + bv       -> Vt[b][c][t]  (f16, TRANSPOSED)
// One block per 128-row x slab; the 128x1024 f16 x tile stays resident in LDS
// (256 KB) while we loop over all 16 N-tiles -> x crosses HBM exactly once.
// Weight chunks are double-buffered async loads.
// ---------------------------------------------------------------------------
__global__ __launch_bounds__(256) void kv_proj_kernel(
    const _Float16* __restrict__ xh, const _Float16* __restrict__ Whk, const float* __restrict__ bk,
    const _Float16* __restrict__ Whv, const float* __restrict__ bv,
    _Float16* __restrict__ Kh, _Float16* __restrict__ Vt) {
  extern __shared__ char smem[];
  _Float16* Xs  = (_Float16*)smem;        // 128*1024 (resident)
  _Float16* Wks = Xs + 128 * 1024;        // 2 x 64*32
  _Float16* Wvs = Wks + 2 * 64 * 32;      // 2 x 64*32

  const int tid  = threadIdx.x;
  const int lane = tid & 31, wvId = tid >> 5;
  const int hi   = lane >> 4, l15 = lane & 15;
  const int m0   = blockIdx.x * 128;
  const int k0   = hi ? 8 : 0;

  {  // resident x slab: 128x1024 f16, async B128 copies
    int row   = tid >> 1;
    int cbase = (tid & 1) * 512;
    const _Float16* src = xh + (size_t)(m0 + row) * CC + cbase;
    _Float16* dst = Xs + row * 1024 + cbase;
    for (int i = 0; i < 64; ++i) cp16_async(dst + i * 8, src + i * 8);
  }
  async_wait0();
  __syncthreads();

  v8f zero = {};
  for (int n0 = 0; n0 < CC; n0 += 64) {
    v8f accK[4], accV[4];
#pragma unroll
    for (int i = 0; i < 4; ++i) { accK[i] = zero; accV[i] = zero; }

    auto issueW = [&](int buf, int kk) {
      int row  = tid >> 2;
      int colq = (tid & 3) * 8;
      cp16_async(Wks + buf * 2048 + row * 32 + colq, Whk + (size_t)(n0 + row) * CC + kk + colq);
      cp16_async(Wvs + buf * 2048 + row * 32 + colq, Whv + (size_t)(n0 + row) * CC + kk + colq);
    };

    issueW(0, 0);
    for (int ci = 0; ci < 32; ++ci) {
      async_wait0();
      __syncthreads();
      if (ci < 31) issueW((ci + 1) & 1, (ci + 1) * 32);
      const int kk = ci * 32, buf = ci & 1;
      v16h a = frag_ld(Xs + (wvId * 16 + l15) * 1024 + kk, k0);
#pragma unroll
      for (int nt = 0; nt < 4; ++nt) {
        v16h bkf = frag_ld(Wks + buf * 2048 + (nt * 16 + l15) * 32, k0);
        accK[nt] = wmma_f16(a, bkf, accK[nt]);
        v16h bvf = frag_ld(Wvs + buf * 2048 + (nt * 16 + l15) * 32, k0);
        accV[nt] = wmma_f16(a, bvf, accV[nt]);
      }
    }

    // Epilogue: bias, RoPE(K), f16 stores.  C/D layout: elem j -> (m=j+8*hi, n=l15).
#pragma unroll
    for (int nt = 0; nt < 4; ++nt) {
      int   c    = n0 + nt * 16 + l15;
      float bkv  = bk[c];
      float bvv  = bv[c];
      int   f    = c & ~1;
      float invf = __expf(-LN10000 * (float)f * (1.0f / CC));
      bool  odd  = (c & 1) != 0;
#pragma unroll
      for (int j = 0; j < 8; ++j) {
        int   row = m0 + wvId * 16 + j + 8 * hi;
        int   bi  = row >> 12;          // T = 4096
        int   t   = row & (TT - 1);
        float kv  = accK[nt][j] + bkv;
        float prt = __shfl_xor(kv, 1, 32);   // partner column (c^1), same row
        float s, cz;
        __sincosf((float)t * invf, &s, &cz);
        float kro = odd ? (kv * cz + prt * s) : (kv * cz - prt * s);
        Kh[(size_t)row * CC + c] = (_Float16)kro;
        Vt[((size_t)bi * CC + c) * TT + t] = (_Float16)(accV[nt][j] + bvv);  // transposed
      }
    }
  }
}

// ---------------------------------------------------------------------------
// Kernel 3: flash attention.  Block = (b, 32-query tile), 256 threads.
// Waves: scores -> (qs = w>>2) x (ts = w&3, 32 t each); O -> same qs, 256-c slice.
// Q resident in LDS; K chunks and (pre-transposed) V chunks double-buffered
// via async global->LDS, overlapped with WMMA.
// ---------------------------------------------------------------------------
__global__ __launch_bounds__(256) void attn_kernel(
    const _Float16* __restrict__ Qh, const _Float16* __restrict__ Kh,
    const _Float16* __restrict__ Vt, _Float16* __restrict__ Oh) {
  extern __shared__ char smem[];
  _Float16* Qs  = (_Float16*)smem;            // 32*1024 (resident)
  _Float16* Ks  = Qs + 32 * 1024;             // 2 x 128*32
  _Float16* Ps  = Ks + 2 * 128 * 32;          // 32*128
  _Float16* Vs  = Ps + 32 * 128;              // 2 x 1024*32  ([c][t], already transposed)
  float* redM   = (float*)(Vs + 2 * 1024 * 32);  // [32][4]
  float* redS   = redM + 128;                    // [32][4]
  float* rowM   = redS + 128;                    // [32]
  float* rowL   = rowM + 32;                     // [32]
  float* rowSc  = rowL + 32;                     // [32]

  const int tid  = threadIdx.x;
  const int lane = tid & 31, wvId = tid >> 5;
  const int hi   = lane >> 4, l15 = lane & 15;
  const int b    = blockIdx.x >> 4;
  const int q0   = (blockIdx.x & 15) * 32;
  const int qs   = wvId >> 2;            // query subtile (16 rows)
  const int ts   = wvId & 3;             // 32-wide t slice for scores
  const int crange = (wvId & 3) * 256;   // 256-wide c slice for O
  const int k0   = hi ? 8 : 0;
  const size_t kbase = (size_t)b * TT * CC;
  const size_t vbase = (size_t)b * CC * TT;

  {  // resident Q tile: 32x1024 f16 (64KB)
    int row = tid >> 3;
    int c0  = (tid & 7) * 128;
    const _Float16* src = Qh + (size_t)(q0 + row) * CC + c0;
    _Float16* dst = Qs + row * 1024 + c0;
    for (int i = 0; i < 16; ++i) cp16_async(dst + i * 8, src + i * 8);
  }
  if (tid < 32) { rowM[tid] = -1e30f; rowL[tid] = 0.0f; }
  async_wait0();
  __syncthreads();

  v8f zero = {};
  v8f accO[16];
#pragma unroll
  for (int i = 0; i < 16; ++i) accO[i] = zero;

  for (int t0 = 0; t0 < TT; t0 += 128) {
    auto issueKs = [&](int buf, int c0) {
      int row  = tid >> 1;
      int colh = (tid & 1) * 16;
      const _Float16* src = Kh + kbase + (size_t)(t0 + row) * CC + c0 + colh;
      _Float16* dst = Ks + buf * (128 * 32) + row * 32 + colh;
      cp16_async(dst, src);
      cp16_async(dst + 8, src + 8);
    };
    auto issueVs = [&](int buf, int kt) {
#pragma unroll
      for (int r = 0; r < 4; ++r) {
        int c = tid + r * 256;
        const _Float16* src = Vt + vbase + (size_t)c * TT + t0 + kt * 32;
        _Float16* dst = Vs + buf * (1024 * 32) + c * 32;
#pragma unroll
        for (int q = 0; q < 4; ++q) cp16_async(dst + q * 8, src + q * 8);
      }
    };

    // ---- scores: S(32q x 128t) = Q . K^T over C, K chunks double-buffered ----
    v8f s0 = zero, s1 = zero;
    issueKs(0, 0);
    for (int ci = 0; ci < 32; ++ci) {
      async_wait0();
      __syncthreads();
      if (ci < 31) issueKs((ci + 1) & 1, (ci + 1) * 32);
      const _Float16* Kb = Ks + (ci & 1) * (128 * 32);
      const int c0 = ci * 32;
      v16h a  = frag_ld(Qs + (qs * 16 + l15) * 1024 + c0, k0);
      v16h b0 = frag_ld(Kb + (ts * 32 + l15) * 32, k0);
      s0 = wmma_f16(a, b0, s0);
      v16h b1 = frag_ld(Kb + (ts * 32 + 16 + l15) * 32, k0);
      s1 = wmma_f16(a, b1, s1);
    }

    // prefetch first V chunk; it overlaps the whole softmax phase
    issueVs(0, 0);

    // ---- online softmax ----
    float mx[8];
#pragma unroll
    for (int j = 0; j < 8; ++j) {
      float m = fmaxf(s0[j], s1[j]);
#pragma unroll
      for (int d = 1; d < 16; d <<= 1) m = fmaxf(m, __shfl_xor(m, d, 32));
      mx[j] = m;
    }
    if (l15 == 0) {
#pragma unroll
      for (int j = 0; j < 8; ++j) redM[(qs * 16 + j + 8 * hi) * 4 + ts] = mx[j];
    }
    __syncthreads();
    if (tid < 32) {
      float m = fmaxf(fmaxf(redM[tid * 4 + 0], redM[tid * 4 + 1]),
                      fmaxf(redM[tid * 4 + 2], redM[tid * 4 + 3]));
      float mo = rowM[tid];
      float mn = fmaxf(mo, m);
      rowSc[tid] = __expf(mo - mn);
      rowM[tid]  = mn;
    }
    __syncthreads();

    float sm[8];
#pragma unroll
    for (int j = 0; j < 8; ++j) {
      int   row = qs * 16 + j + 8 * hi;
      float mn  = rowM[row];
      float p0  = __expf(s0[j] - mn);
      float p1  = __expf(s1[j] - mn);
      Ps[row * 128 + ts * 32 + l15]      = (_Float16)p0;
      Ps[row * 128 + ts * 32 + 16 + l15] = (_Float16)p1;
      float sum = p0 + p1;
#pragma unroll
      for (int d = 1; d < 16; d <<= 1) sum += __shfl_xor(sum, d, 32);
      sm[j] = sum;
    }
    if (l15 == 0) {
#pragma unroll
      for (int j = 0; j < 8; ++j) redS[(qs * 16 + j + 8 * hi) * 4 + ts] = sm[j];
    }
    __syncthreads();
    if (tid < 32) {
      rowL[tid] = rowL[tid] * rowSc[tid] +
                  redS[tid * 4 + 0] + redS[tid * 4 + 1] + redS[tid * 4 + 2] + redS[tid * 4 + 3];
    }
    {  // rescale O accumulators by exp(m_old - m_new)
      float sc[8];
#pragma unroll
      for (int j = 0; j < 8; ++j) sc[j] = rowSc[qs * 16 + j + 8 * hi];
#pragma unroll
      for (int ct = 0; ct < 16; ++ct) {
#pragma unroll
        for (int j = 0; j < 8; ++j) accO[ct][j] *= sc[j];
      }
    }

    // ---- O += P @ V, V chunks double-buffered (already [c][t] in memory) ----
    for (int kt = 0; kt < 4; ++kt) {
      async_wait0();
      __syncthreads();
      if (kt < 3) issueVs((kt + 1) & 1, kt + 1);
      const _Float16* Vb = Vs + (kt & 1) * (1024 * 32);
      v16h a = frag_ld(Ps + (qs * 16 + l15) * 128 + kt * 32, k0);
#pragma unroll
      for (int ct = 0; ct < 16; ++ct) {
        v16h bb = frag_ld(Vb + (crange + ct * 16 + l15) * 32, k0);
        accO[ct] = wmma_f16(a, bb, accO[ct]);
      }
    }
  }

  // ---- normalize and store O (f16) ----
  __syncthreads();
  float inv[8];
#pragma unroll
  for (int j = 0; j < 8; ++j) inv[j] = 1.0f / rowL[qs * 16 + j + 8 * hi];
#pragma unroll
  for (int ct = 0; ct < 16; ++ct) {
#pragma unroll
    for (int j = 0; j < 8; ++j) {
      int qg = q0 + qs * 16 + j + 8 * hi;
      int cg = crange + ct * 16 + l15;
      Oh[((size_t)b * QQ + qg) * CC + cg] = (_Float16)(accO[ct][j] * inv[j]);
    }
  }
}

// ---------------------------------------------------------------------------
// Kernel 4: out = O @ Wf.T + bf  (f32 out, (B*QS, C)); double-buffered async.
// ---------------------------------------------------------------------------
__global__ __launch_bounds__(256) void out_proj_kernel(
    const _Float16* __restrict__ Oh, const _Float16* __restrict__ Whf,
    const float* __restrict__ bf, float* __restrict__ out) {
  __shared__ _Float16 Os[2 * 128 * 32];
  __shared__ _Float16 Wfs[2 * 64 * 32];

  const int tid  = threadIdx.x;
  const int lane = tid & 31, wvId = tid >> 5;
  const int hi   = lane >> 4, l15 = lane & 15;
  const int m0   = blockIdx.x * 128;
  const int n0   = blockIdx.y * 64;
  const int k0   = hi ? 8 : 0;

  v8f zero = {};
  v8f acc[4];
#pragma unroll
  for (int i = 0; i < 4; ++i) acc[i] = zero;

  auto issue = [&](int buf, int kk) {
    {
      int row  = tid >> 1;
      int colh = (tid & 1) * 16;
      const _Float16* src = Oh + (size_t)(m0 + row) * CC + kk + colh;
      _Float16* dst = Os + buf * 4096 + row * 32 + colh;
      cp16_async(dst, src);
      cp16_async(dst + 8, src + 8);
    }
    {
      int row  = tid >> 2;
      int colq = (tid & 3) * 8;
      cp16_async(Wfs + buf * 2048 + row * 32 + colq, Whf + (size_t)(n0 + row) * CC + kk + colq);
    }
  };

  issue(0, 0);
  for (int ci = 0; ci < 32; ++ci) {
    async_wait0();
    __syncthreads();
    if (ci < 31) issue((ci + 1) & 1, (ci + 1) * 32);
    const int buf = ci & 1;
    v16h a = frag_ld(Os + buf * 4096 + (wvId * 16 + l15) * 32, k0);
#pragma unroll
    for (int nt = 0; nt < 4; ++nt) {
      v16h bb = frag_ld(Wfs + buf * 2048 + (nt * 16 + l15) * 32, k0);
      acc[nt] = wmma_f16(a, bb, acc[nt]);
    }
  }

#pragma unroll
  for (int nt = 0; nt < 4; ++nt) {
    int   c   = n0 + nt * 16 + l15;
    float bfv = bf[c];
#pragma unroll
    for (int j = 0; j < 8; ++j) {
      int row = m0 + wvId * 16 + j + 8 * hi;
      out[(size_t)row * CC + c] = acc[nt][j] + bfv;
    }
  }
}

// ---------------------------------------------------------------------------
extern "C" void kernel_launch(void* const* d_in, const int* in_sizes, int n_in,
                              void* d_out, int out_size, void* d_ws, size_t ws_size,
                              hipStream_t stream) {
  const float* x  = (const float*)d_in[0];
  const float* Wq = (const float*)d_in[1];
  const float* Wk = (const float*)d_in[2];
  const float* bk = (const float*)d_in[3];
  const float* Wv = (const float*)d_in[4];
  const float* bv = (const float*)d_in[5];
  const float* Wf = (const float*)d_in[6];
  const float* bf = (const float*)d_in[7];
  float* out = (float*)d_out;

  char* w = (char*)d_ws;
  const size_t xBytes  = (size_t)BB * TT * CC * 2;   // 64 MB
  const size_t qBytes  = (size_t)QQ * CC * 2;        // 1 MB
  const size_t oBytes  = (size_t)BB * QQ * CC * 2;   // 8 MB
  const size_t wBytes  = (size_t)CC * CC * 2;        // 2 MB
  _Float16* xh  = (_Float16*)w;                       w += xBytes;
  _Float16* Qh  = (_Float16*)w;                       w += qBytes;
  _Float16* Kh  = (_Float16*)w;                       w += xBytes;
  _Float16* Vt  = (_Float16*)w;                       w += xBytes;
  _Float16* Oh  = (_Float16*)w;                       w += oBytes;
  _Float16* Whk = (_Float16*)w;                       w += wBytes;
  _Float16* Whv = (_Float16*)w;                       w += wBytes;
  _Float16* Whf = (_Float16*)w;                       w += wBytes;

  // f32 -> f16 prep
  cvt_f16_kernel<<<(BB * TT * CC / 8) / 256, 256, 0, stream>>>(x, xh, BB * TT * CC / 8);
  cvt_f16_kernel<<<(CC * CC / 8) / 256, 256, 0, stream>>>(Wk, Whk, CC * CC / 8);
  cvt_f16_kernel<<<(CC * CC / 8) / 256, 256, 0, stream>>>(Wv, Whv, CC * CC / 8);
  cvt_f16_kernel<<<(CC * CC / 8) / 256, 256, 0, stream>>>(Wf, Whf, CC * CC / 8);
  rope_q_kernel<<<(QQ * (CC / 2)) / 256, 256, 0, stream>>>(Wq, Qh);

  size_t smemKV = (size_t)(128 * 1024 + 4 * 64 * 32) * 2;  // ~272 KB (WGP LDS = 320 KB)
  kv_proj_kernel<<<dim3((BB * TT) / 128), 256, smemKV, stream>>>(xh, Whk, bk, Whv, bv, Kh, Vt);

  size_t smemA = (size_t)(32 * 1024 + 2 * 128 * 32 + 32 * 128 + 2 * 1024 * 32) * 2 +
                 (size_t)(128 + 128 + 32 + 32 + 32) * 4;   // ~218 KB
  attn_kernel<<<dim3(BB * (QQ / 32)), 256, smemA, stream>>>(Qh, Kh, Vt, Oh);

  out_proj_kernel<<<dim3((BB * QQ) / 128, CC / 64), 256, 0, stream>>>(Oh, Whf, bf, out);
}
